// SoftMergingMoE_9294309228787
// MI455X (gfx1250) — matrix operand order repaired and checked by reading, not compiled, not run on hardware
//
#include <hip/hip_runtime.h>
#include <stdint.h>

#define HDIM 1024
#define EXP  8
#define FDIM 2730
#define LSEG 256
#define GRP  32
#define FPAD 2816   // 22 * 128, covers FDIM with zero padding

typedef __attribute__((ext_vector_type(16))) __bf16 v16bf;
typedef __attribute__((ext_vector_type(8)))  float  v8f;

union Frag {
  v16bf v;
  uint4 q[2];
};

__device__ __forceinline__ unsigned short f32_to_bf16(float f) {
  union { float f; unsigned int u; } c; c.f = f;
  unsigned int u = c.u;
  return (unsigned short)((u + 0x7FFFu + ((u >> 16) & 1u)) >> 16); // RNE
}

__device__ __forceinline__ v8f vzero8() {
  v8f z;
#pragma unroll
  for (int i = 0; i < 8; ++i) z[i] = 0.0f;
  return z;
}

// ---------------------------------------------------------------------------
// Kernel 1: per-(g,h) mean over the L=256 segment rows.
// seg[l,g,h] = hidden_flat[(l*32+g)*H + h]
// ---------------------------------------------------------------------------
__global__ __launch_bounds__(256)
void moe_mean_kernel(const float* __restrict__ hs, float* __restrict__ mean) {
  int idx = blockIdx.x * 256 + threadIdx.x;   // 0 .. 32*1024-1
  int g = idx >> 10;
  int h = idx & 1023;
  const float* p = hs + (size_t)g * HDIM + h;
  float s = 0.0f;
#pragma unroll 4
  for (int l = 0; l < LSEG; ++l) s += p[(size_t)l * (GRP * HDIM)];
  mean[idx] = s * (1.0f / LSEG);
}

// ---------------------------------------------------------------------------
// Kernel 2: router logits -> softmax -> roll trick -> ew[32][8]
// ---------------------------------------------------------------------------
__global__ __launch_bounds__(256)
void moe_router_kernel(const float* __restrict__ mean,
                       const float* __restrict__ rw,
                       float* __restrict__ ew) {
  __shared__ float sm[GRP][EXP];
  __shared__ float pr[GRP][EXP];
  int t = threadIdx.x;
  int g = t >> 3, e = t & 7;
  const float* m = mean + (size_t)g * HDIM;
  const float* w = rw + (size_t)e * HDIM;
  float acc = 0.0f;
  for (int h = 0; h < HDIM; h += 4) {
    acc = fmaf(m[h + 0], w[h + 0], acc);
    acc = fmaf(m[h + 1], w[h + 1], acc);
    acc = fmaf(m[h + 2], w[h + 2], acc);
    acc = fmaf(m[h + 3], w[h + 3], acc);
  }
  sm[g][e] = acc;
  __syncthreads();
  float mx = sm[g][0];
#pragma unroll
  for (int i = 1; i < EXP; ++i) mx = fmaxf(mx, sm[g][i]);
  float ex = __expf(acc - mx);
  pr[g][e] = ex;
  __syncthreads();
  float den = 0.0f;
#pragma unroll
  for (int i = 0; i < EXP; ++i) den += pr[g][i];
  float val = ex / den;
  sm[g][e] = val;                 // normalized probabilities
  __syncthreads();
  float out;
  if (g < GRP / 2) {
    out = val;                    // first batch row restored
  } else {
    int flat = g * EXP + e - 1;   // roll(flat,1): new[i] = old[i-1], i>=128
    out = (&sm[0][0])[flat];
  }
  ew[g * EXP + e] = out;
}

// ---------------------------------------------------------------------------
// Kernel 3: fused  t = silu(x @ mw1) * (x @ mw3)   (bf16 out, F padded)
// block = (g, ftile of 128, ltile of 128); 256 threads = 8 waves (4M x 2N)
// ---------------------------------------------------------------------------
__global__ __launch_bounds__(256)
void moe_gemm1_kernel(const float* __restrict__ hs,
                      const float* __restrict__ w1,
                      const float* __restrict__ w3,
                      const float* __restrict__ ew,
                      unsigned short* __restrict__ t) {
  __shared__ __align__(16) unsigned short As[128][40];   // [row][k]
  __shared__ __align__(16) unsigned short W1s[128][40];  // [n][k]
  __shared__ __align__(16) unsigned short W3s[128][40];  // [n][k]
  __shared__ float ews[EXP];

  const int g  = blockIdx.x;
  const int f0 = blockIdx.y * 128;
  const int r0 = blockIdx.z * 128;
  const int tid  = threadIdx.x;
  const int lane = tid & 31;
  const int wave = tid >> 5;
  const int wm = wave >> 1;   // 0..3 : 32-row band
  const int wn = wave & 1;    // 0..1 : 64-col band

  if (tid < EXP) ews[tid] = ew[g * EXP + tid];
  __syncthreads();
  float es[EXP];
#pragma unroll
  for (int e = 0; e < EXP; ++e) es[e] = ews[e];

  v8f acc1[2][4], acc3[2][4];
#pragma unroll
  for (int m = 0; m < 2; ++m)
#pragma unroll
    for (int n = 0; n < 4; ++n) { acc1[m][n] = vzero8(); acc3[m][n] = vzero8(); }

  const int ka  = tid & 31;     // A staging: k
  const int ra  = tid >> 5;     // A staging: row group
  const int nb  = tid & 127;    // B staging: n (column of F tile)
  const int kb0 = tid >> 7;     // B staging: k parity
  const bool nok = (f0 + nb) < FDIM;

  const int mr  = lane & 15;
  const int ksA = (lane < 16) ? 0 : 8;    // A frag K base (elements)
  const int ksB = (lane < 16) ? 0 : 8;    // B frag K base (elements)

  for (int h0 = 0; h0 < HDIM; h0 += 32) {
    // ---- stage x tile -> bf16 LDS (row-major [row][k]) ----
#pragma unroll
    for (int r = ra; r < 128; r += 8) {
      const float* src = &hs[((size_t)((r0 + r) * GRP + g)) * HDIM + h0 + ka];
      if (h0 + 32 < HDIM) __builtin_prefetch(src + 32, 0, 0);  // global_prefetch_b8
      As[r][ka] = f32_to_bf16(*src);
    }
    // ---- merge 8 experts of w1/w3 -> bf16 LDS (transposed [n][k]) ----
    for (int k = kb0; k < 32; k += 2) {
      float a1 = 0.0f, a3 = 0.0f;
      if (nok) {
        size_t base = ((size_t)(h0 + k)) * FDIM + (f0 + nb);
#pragma unroll
        for (int e = 0; e < EXP; ++e) {
          a1 = fmaf(es[e], w1[(size_t)e * HDIM * FDIM + base], a1);
          a3 = fmaf(es[e], w3[(size_t)e * HDIM * FDIM + base], a3);
        }
      }
      W1s[nb][k] = f32_to_bf16(a1);
      W3s[nb][k] = f32_to_bf16(a3);
    }
    __syncthreads();

    // ---- WMMA: 2 M-subtiles x 4 N-subtiles x 2 matrices ----
    Frag a[2];
#pragma unroll
    for (int m = 0; m < 2; ++m) {
      const unsigned short* pa = &As[wm * 32 + m * 16 + mr][0];
      a[m].q[0] = *(const uint4*)(pa + ksA);        // K ksA..ksA+7
      a[m].q[1] = *(const uint4*)(pa + ksA + 16);   // K ksA+16..ksA+23
    }
#pragma unroll
    for (int n = 0; n < 4; ++n) {
      Frag b1, b3;
      const unsigned short* p1 = &W1s[wn * 64 + n * 16 + mr][0];
      const unsigned short* p3 = &W3s[wn * 64 + n * 16 + mr][0];
      b1.q[0] = *(const uint4*)(p1 + ksB);
      b1.q[1] = *(const uint4*)(p1 + ksB + 16);
      b3.q[0] = *(const uint4*)(p3 + ksB);
      b3.q[1] = *(const uint4*)(p3 + ksB + 16);
#pragma unroll
      for (int m = 0; m < 2; ++m) {
        acc1[m][n] = __builtin_amdgcn_wmma_f32_16x16x32_bf16(
            false, a[m].v, false, b1.v, (short)0, acc1[m][n], false, false);
        acc3[m][n] = __builtin_amdgcn_wmma_f32_16x16x32_bf16(
            false, a[m].v, false, b3.v, (short)0, acc3[m][n], false, false);
      }
    }
    __syncthreads();
  }

  // ---- epilogue: t = silu(o1) * o3, bf16. Padded F region is exactly 0. ----
  const int nloc = lane & 15;
  const int mhi  = (lane >> 4) << 3;   // 0 or 8
#pragma unroll
  for (int m = 0; m < 2; ++m)
#pragma unroll
    for (int n = 0; n < 4; ++n)
#pragma unroll
      for (int r = 0; r < 8; ++r) {
        float o1 = acc1[m][n][r];
        float o3 = acc3[m][n][r];
        float sig = 1.0f / (1.0f + __expf(-o1));
        float tv = o1 * sig * o3;
        int lrow = r0 + wm * 32 + m * 16 + r + mhi;
        int fcol = f0 + wn * 64 + n * 16 + nloc;
        t[((size_t)(g * LSEG + lrow)) * FPAD + fcol] = f32_to_bf16(tv);
      }
}

// ---------------------------------------------------------------------------
// Kernel 4: o2 = t @ mw2  (K = FPAD, zero-padded), f32 out straight to d_out
// block = (g, htile of 128, ltile of 128)
// ---------------------------------------------------------------------------
__global__ __launch_bounds__(256)
void moe_gemm2_kernel(const unsigned short* __restrict__ t,
                      const float* __restrict__ w2,
                      const float* __restrict__ ew,
                      float* __restrict__ out) {
  __shared__ __align__(16) unsigned short As[128][40];
  __shared__ __align__(16) unsigned short W2s[128][40];
  __shared__ float ews[EXP];

  const int g  = blockIdx.x;
  const int h0 = blockIdx.y * 128;
  const int r0 = blockIdx.z * 128;
  const int tid  = threadIdx.x;
  const int lane = tid & 31;
  const int wave = tid >> 5;
  const int wm = wave >> 1;
  const int wn = wave & 1;

  if (tid < EXP) ews[tid] = ew[g * EXP + tid];
  __syncthreads();
  float es[EXP];
#pragma unroll
  for (int e = 0; e < EXP; ++e) es[e] = ews[e];

  v8f acc[2][4];
#pragma unroll
  for (int m = 0; m < 2; ++m)
#pragma unroll
    for (int n = 0; n < 4; ++n) acc[m][n] = vzero8();

  const int ka  = tid & 31;
  const int ra  = tid >> 5;
  const int nb  = tid & 127;
  const int kb0 = tid >> 7;

  const int mr  = lane & 15;
  const int ksA = (lane < 16) ? 0 : 8;
  const int ksB = (lane < 16) ? 0 : 8;

  for (int f0k = 0; f0k < FPAD; f0k += 32) {
    // ---- stage t tile (already bf16) ----
#pragma unroll
    for (int r = ra; r < 128; r += 8)
      As[r][ka] = t[((size_t)(g * LSEG + r0 + r)) * FPAD + f0k + ka];
    // ---- merge 8 experts of w2 -> [n][k], guard K tail ----
    for (int k = kb0; k < 32; k += 2) {
      float a2 = 0.0f;
      int fi = f0k + k;
      if (fi < FDIM) {
#pragma unroll
        for (int e = 0; e < EXP; ++e)
          a2 = fmaf(es[e], w2[((size_t)e * FDIM + fi) * HDIM + h0 + nb], a2);
      }
      W2s[nb][k] = f32_to_bf16(a2);
    }
    __syncthreads();

    Frag a[2];
#pragma unroll
    for (int m = 0; m < 2; ++m) {
      const unsigned short* pa = &As[wm * 32 + m * 16 + mr][0];
      a[m].q[0] = *(const uint4*)(pa + ksA);
      a[m].q[1] = *(const uint4*)(pa + ksA + 16);
    }
#pragma unroll
    for (int n = 0; n < 4; ++n) {
      Frag b;
      const unsigned short* pb = &W2s[wn * 64 + n * 16 + mr][0];
      b.q[0] = *(const uint4*)(pb + ksB);
      b.q[1] = *(const uint4*)(pb + ksB + 16);
#pragma unroll
      for (int m = 0; m < 2; ++m)
        acc[m][n] = __builtin_amdgcn_wmma_f32_16x16x32_bf16(
            false, a[m].v, false, b.v, (short)0, acc[m][n], false, false);
    }
    __syncthreads();
  }

  // ---- epilogue: out_flat[(l*32+g)*H + h] = o2 ----
  const int nloc = lane & 15;
  const int mhi  = (lane >> 4) << 3;
#pragma unroll
  for (int m = 0; m < 2; ++m)
#pragma unroll
    for (int n = 0; n < 4; ++n)
#pragma unroll
      for (int r = 0; r < 8; ++r) {
        int lrow = r0 + wm * 32 + m * 16 + r + mhi;
        int hcol = h0 + wn * 64 + n * 16 + nloc;
        out[((size_t)(lrow * GRP + g)) * HDIM + hcol] = acc[m][n][r];
      }
}

// ---------------------------------------------------------------------------
extern "C" void kernel_launch(void* const* d_in, const int* in_sizes, int n_in,
                              void* d_out, int out_size, void* d_ws, size_t ws_size,
                              hipStream_t stream) {
  const float* hs = (const float*)d_in[0];   // [4096, 2, 1024]
  const float* rw = (const float*)d_in[1];   // [8, 1024]
  const float* w1 = (const float*)d_in[2];   // [8, 1024, 2730]
  const float* w2 = (const float*)d_in[3];   // [8, 2730, 1024]
  const float* w3 = (const float*)d_in[4];   // [8, 1024, 2730]
  float* out = (float*)d_out;

  // workspace layout: ew (1 KB) | mean (128 KB) | t bf16 [32][256][2816] (~46 MB)
  float* ws_f = (float*)d_ws;
  float* ew   = ws_f;                                        // 256 floats
  float* mean = ws_f + 256;                                  // 32768 floats
  unsigned short* t = (unsigned short*)((char*)d_ws + 262144);

  moe_mean_kernel<<<dim3((GRP * HDIM) / 256), 256, 0, stream>>>(hs, mean);
  moe_router_kernel<<<dim3(1), 256, 0, stream>>>(mean, rw, ew);
  moe_gemm1_kernel<<<dim3(GRP, FPAD / 128, LSEG / 128), 256, 0, stream>>>(hs, w1, w3, ew, t);
  moe_gemm2_kernel<<<dim3(GRP, HDIM / 128, LSEG / 128), 256, 0, stream>>>(t, w2, ew, out);
}
